// transform_net_84353157693736
// MI455X (gfx1250) — compile-verified, matrix operand order checked by reading, not run on hardware
//
#include <hip/hip_runtime.h>

typedef __attribute__((ext_vector_type(16))) _Float16 v16h;
typedef __attribute__((ext_vector_type(8)))  _Float16 v8h;
typedef __attribute__((ext_vector_type(2)))  _Float16 v2h;
typedef __attribute__((ext_vector_type(8)))  float    v8f;

union Frag16 { v16h v; v8h h[2]; };

#define EPSV 1e-5f
#define NEG_INF -3.0e38f

// ---------------------------------------------------------------- utilities
__global__ void zero_kernel(float* __restrict__ p, int n) {
  int i = blockIdx.x * blockDim.x + threadIdx.x;
  if (i < n) p[i] = 0.0f;
}

__global__ void sq_kernel(const float* __restrict__ pc, float* __restrict__ sq) {
  int i = blockIdx.x * blockDim.x + threadIdx.x;  // 32768 = 16*2048
  int b = i >> 11, n = i & 2047;
  float a0 = pc[(b * 3 + 0) * 2048 + n];
  float a1 = pc[(b * 3 + 1) * 2048 + n];
  float a2 = pc[(b * 3 + 2) * 2048 + n];
  sq[i] = a0 * a0 + a1 * a1 + a2 * a2;
}

// ---------------------------------------------------------------- kNN top-20
__global__ __launch_bounds__(256) void knn_kernel(const float* __restrict__ pc,
                                                  const float* __restrict__ sq,
                                                  int* __restrict__ idx) {
  const int N = 2048, K = 20;
  int row = blockIdx.x;          // b*2048+n
  int b = row >> 11, n = row & 2047;
  int t = threadIdx.x;
  __shared__ float rv[256];
  __shared__ int   ri[256];
  __shared__ int   s_win;
  float c0 = pc[(b * 3 + 0) * N + n];
  float c1 = pc[(b * 3 + 1) * N + n];
  float c2 = pc[(b * 3 + 2) * N + n];
  float sqn = sq[row];
  float d[8];
  int m0 = t * 8;
#pragma unroll
  for (int j = 0; j < 8; ++j) {
    int m = m0 + j;
    float inner = c0 * pc[(b * 3 + 0) * N + m] + c1 * pc[(b * 3 + 1) * N + m] +
                  c2 * pc[(b * 3 + 2) * N + m];
    d[j] = 2.0f * inner - sqn - sq[b * N + m];
  }
  for (int it = 0; it < K; ++it) {
    float bv = NEG_INF;
    int bi = 0;
#pragma unroll
    for (int j = 0; j < 8; ++j) {
      if (d[j] > bv) { bv = d[j]; bi = m0 + j; }   // ascending j -> lowest index on tie
    }
    rv[t] = bv; ri[t] = bi;
    __syncthreads();
    for (int s = 128; s > 0; s >>= 1) {
      if (t < s) {
        if (rv[t + s] > rv[t] || (rv[t + s] == rv[t] && ri[t + s] < ri[t])) {
          rv[t] = rv[t + s]; ri[t] = ri[t + s];
        }
      }
      __syncthreads();
    }
    if (t == 0) { idx[row * K + it] = ri[0]; s_win = ri[0]; }
    __syncthreads();
    int w = s_win;
    if ((w >> 3) == t) d[w & 7] = NEG_INF;
    __syncthreads();
  }
}

// ---------------------------------------------------------------- BN1 stats
__global__ __launch_bounds__(256) void bn1_stats_kernel(const float* __restrict__ pc,
                                                        const int* __restrict__ idx,
                                                        const float* __restrict__ W1,
                                                        float* __restrict__ bn1) {
  __shared__ float s_sum[64], s_sq[64];
  int t = threadIdx.x;
  if (t < 64) { s_sum[t] = 0.f; s_sq[t] = 0.f; }
  __syncthreads();
  int r = blockIdx.x * 256 + t;  // < 655360
  int bn = r / 20;
  int n = bn & 2047, b = bn >> 11;
  int m = idx[r];
  float e[6];
#pragma unroll
  for (int c = 0; c < 3; ++c) {
    float cn = pc[(b * 3 + c) * 2048 + n];
    float nm = pc[(b * 3 + c) * 2048 + m];
    e[c] = cn; e[c + 3] = nm - cn;
  }
  for (int o = 0; o < 64; ++o) {
    float h = 0.f;
#pragma unroll
    for (int c = 0; c < 6; ++c) h += e[c] * W1[c * 64 + o];
    atomicAdd(&s_sum[o], h);
    atomicAdd(&s_sq[o], h * h);
  }
  __syncthreads();
  if (t < 64) { atomicAdd(&bn1[t], s_sum[t]); atomicAdd(&bn1[64 + t], s_sq[t]); }
}

// blk layout: [sum(nch)][sumsq(nch)][scale(nch)][shift(nch)]
__global__ void bn_finalize_kernel(float* __restrict__ blk, const float* __restrict__ g,
                                   const float* __restrict__ bb, int nch, float invcount) {
  int t = blockIdx.x * blockDim.x + threadIdx.x;
  if (t >= nch) return;
  float mean = blk[t] * invcount;
  float var = blk[nch + t] * invcount - mean * mean;
  float sc = g[t] * rsqrtf(var + EPSV);
  blk[2 * nch + t] = sc;
  blk[3 * nch + t] = bb[t] - mean * sc;
}

// ---------------------------------------------------------------- conv2 (WMMA)
// W is stored K-swizzled in LDS: s_W[(ks*128 + col)*32 + kl] = W[ks*32+kl][col]
// so every A/B fragment is two contiguous 16B (v8h) LDS loads.
// mode 0: accumulate BN2 stats.  mode 1: apply BN2+ReLU, max over k -> h2max.
__global__ __launch_bounds__(256) void conv2_kernel(const float* __restrict__ pc,
                                                    const int* __restrict__ idx,
                                                    const float* __restrict__ W1,
                                                    const float* __restrict__ bn1,
                                                    const float* __restrict__ W2,
                                                    float* __restrict__ bn2,
                                                    float* __restrict__ h2max, int mode) {
  __shared__ float                 s_edge[80 * 6];
  __shared__ __align__(16) _Float16 s_X[80 * 64];    // [row][k]
  __shared__ __align__(16) _Float16 s_W[64 * 128];   // K-swizzled (see above)
  __shared__ __align__(16) _Float16 s_H[128 * 80];   // [col][row] (transposed)
  __shared__ float                 s_stats[256];
  int t = threadIdx.x;
  int r0 = blockIdx.x * 80;
  if (t < 128) { s_stats[t] = 0.f; s_stats[128 + t] = 0.f; }
  // stage edge features for 80 rows
  for (int e = t; e < 480; e += 256) {
    int i = e / 6, c = e % 6;
    int r = r0 + i;
    int bn = r / 20;
    int n = bn & 2047, b = bn >> 11;
    int cc = (c < 3) ? c : (c - 3);
    float cn = pc[(b * 3 + cc) * 2048 + n];
    float v = (c < 3) ? cn : (pc[(b * 3 + cc) * 2048 + idx[r]] - cn);
    s_edge[i * 6 + c] = v;
  }
  __syncthreads();
  // recompute h1 -> BN1 -> ReLU -> f16 activations (packed pair stores)
  const float* sc1 = bn1 + 128;
  const float* sh1 = bn1 + 192;
  for (int e = t; e < 2560; e += 256) {
    int i = e >> 5, op = (e & 31) * 2;
    v2h xv;
#pragma unroll
    for (int u = 0; u < 2; ++u) {
      int o = op + u;
      float h = 0.f;
#pragma unroll
      for (int c = 0; c < 6; ++c) h += s_edge[i * 6 + c] * W1[c * 64 + o];
      h = h * sc1[o] + sh1[o];
      xv[u] = (_Float16)fmaxf(h, 0.f);
    }
    *(v2h*)&s_X[i * 64 + op] = xv;
  }
  // W2 (64x128) -> K-swizzled f16
  for (int e = t; e < 4096; e += 256) {
    int col = e & 127, kr = (e >> 7) * 2;         // kr: 0,2,..,62
    int ks = kr >> 5, kl = kr & 31;
    v2h wv;
    wv[0] = (_Float16)W2[kr * 128 + col];
    wv[1] = (_Float16)W2[(kr + 1) * 128 + col];
    *(v2h*)&s_W[((ks * 128 + col) << 5) + kl] = wv;
  }
  __syncthreads();

  int lane = t & 31, wv = t >> 5;
  int l15 = lane & 15, hi = lane >> 4;
  int ncol = wv * 16 + l15;
  float sc2v = 0.f, sh2v = 0.f;
  if (mode == 1) { sc2v = bn2[256 + ncol]; sh2v = bn2[384 + ncol]; }
  float fsum = 0.f, fsq = 0.f;

  const v8h* xp = (const v8h*)s_X;
  const v8h* wp = (const v8h*)s_W;
  // B fragments are tile-row invariant: load once per wave
  Frag16 b0u, b1u;
  int wb0 = (ncol << 2) + hi;            // ks=0: (0*128+ncol)*32/8 + hi
  int wb1 = ((128 + ncol) << 2) + hi;    // ks=1
  b0u.h[0] = wp[wb0]; b0u.h[1] = wp[wb0 + 2];
  b1u.h[0] = wp[wb1]; b1u.h[1] = wp[wb1 + 2];

  for (int mt = 0; mt < 5; ++mt) {
    Frag16 a0u, a1u;
    int ab = (mt * 16 + l15) * 8 + hi;   // row*64/8 + hi
    a0u.h[0] = xp[ab];     a0u.h[1] = xp[ab + 2];
    a1u.h[0] = xp[ab + 4]; a1u.h[1] = xp[ab + 6];
    v8f acc = {};
    acc = __builtin_amdgcn_wmma_f32_16x16x32_f16(false, a0u.v, false, b0u.v, (short)0, acc, false, false);
    acc = __builtin_amdgcn_wmma_f32_16x16x32_f16(false, a1u.v, false, b1u.v, (short)0, acc, false, false);
    if (mode == 0) {
#pragma unroll
      for (int v = 0; v < 8; ++v) { float x = acc[v]; fsum += x; fsq += x * x; }
    } else {
      v8h hv;
#pragma unroll
      for (int v = 0; v < 8; ++v) hv[v] = (_Float16)fmaxf(acc[v] * sc2v + sh2v, 0.f);
      *(v8h*)&s_H[ncol * 80 + mt * 16 + 8 * hi] = hv;   // one b128 store
    }
  }

  if (mode == 0) {
    atomicAdd(&s_stats[ncol], fsum);
    atomicAdd(&s_stats[128 + ncol], fsq);
    __syncthreads();
    if (t < 128) { atomicAdd(&bn2[t], s_stats[t]); atomicAdd(&bn2[128 + t], s_stats[128 + t]); }
  } else {
    __syncthreads();
    for (int e = t; e < 512; e += 256) {
      int g = e >> 7, col = e & 127;
      float mx = 0.f;
#pragma unroll
      for (int j = 0; j < 20; ++j) mx = fmaxf(mx, (float)s_H[col * 80 + g * 20 + j]);
      h2max[(blockIdx.x * 4 + g) * 128 + col] = mx;
    }
  }
}

// ---------------------------------------------------------------- conv3 (WMMA)
// mode 0: BN3 stats.  mode 1: BN3+ReLU, atomic max over N -> h3max[b][1024].
__global__ __launch_bounds__(256) void conv3_kernel(const float* __restrict__ h2max,
                                                    const float* __restrict__ W3,
                                                    float* __restrict__ bn3,
                                                    float* __restrict__ h3max, int mode) {
  __shared__ __align__(16) _Float16 s_X[64 * 128];   // [row][k]
  __shared__ __align__(16) _Float16 s_W[128 * 128];  // K-swizzled
  __shared__ float                  s_stats[256];
  int t = threadIdx.x;
  if (t < 128) { s_stats[t] = 0.f; s_stats[128 + t] = 0.f; }
  int r0 = blockIdx.x * 64;
  int n0 = blockIdx.y * 128;
  for (int e = t; e < 4096; e += 256) {
    v2h xv;
    xv[0] = (_Float16)h2max[r0 * 128 + e * 2];
    xv[1] = (_Float16)h2max[r0 * 128 + e * 2 + 1];
    *(v2h*)&s_X[e * 2] = xv;
  }
  for (int e = t; e < 8192; e += 256) {
    int col = e & 127, kr = (e >> 7) * 2;          // kr: 0,2,..,126
    int ks = kr >> 5, kl = kr & 31;
    v2h wv;
    wv[0] = (_Float16)W3[kr * 1024 + n0 + col];
    wv[1] = (_Float16)W3[(kr + 1) * 1024 + n0 + col];
    *(v2h*)&s_W[((ks * 128 + col) << 5) + kl] = wv;
  }
  __syncthreads();

  int lane = t & 31, wvid = t >> 5;
  int l15 = lane & 15, hi = lane >> 4;
  int b = r0 >> 11;
  const v8h* xp = (const v8h*)s_X;
  const v8h* wp = (const v8h*)s_W;
  for (int job = wvid; job < 32; job += 8) {
    int mt = job >> 3, nt = job & 7;
    int ntcol = nt * 16 + l15;
    v8f acc = {};
#pragma unroll
    for (int ks = 0; ks < 4; ++ks) {
      Frag16 au, bu;
      int ab = (mt * 16 + l15) * 16 + ks * 4 + hi;   // row*128/8 + ks*32/8 + hi
      au.h[0] = xp[ab];     au.h[1] = xp[ab + 2];
      int wb = ((ks * 128 + ntcol) << 2) + hi;
      bu.h[0] = wp[wb];     bu.h[1] = wp[wb + 2];
      acc = __builtin_amdgcn_wmma_f32_16x16x32_f16(false, au.v, false, bu.v, (short)0, acc, false, false);
    }
    if (mode == 0) {
      float fs = 0.f, fq = 0.f;
#pragma unroll
      for (int v = 0; v < 8; ++v) { float x = acc[v]; fs += x; fq += x * x; }
      atomicAdd(&s_stats[ntcol], fs);
      atomicAdd(&s_stats[128 + ntcol], fq);
    } else {
      int gcol = n0 + ntcol;
      float sc = bn3[2048 + gcol], sh = bn3[3072 + gcol];
      float mx = 0.f;
#pragma unroll
      for (int v = 0; v < 8; ++v) mx = fmaxf(mx, fmaxf(acc[v] * sc + sh, 0.f));
      atomicMax((int*)&h3max[b * 1024 + gcol], __float_as_int(mx));  // mx >= 0: int order ok
    }
  }
  if (mode == 0) {
    __syncthreads();
    if (t < 128) {
      atomicAdd(&bn3[n0 + t], s_stats[t]);
      atomicAdd(&bn3[1024 + n0 + t], s_stats[128 + t]);
    }
  }
}

// ---------------------------------------------------------------- tiny FC tail
__global__ void fc_kernel(const float* __restrict__ X, const float* __restrict__ W,
                          float* __restrict__ Y, int rows, int in, int out) {
  int i = blockIdx.x * blockDim.x + threadIdx.x;
  if (i >= rows * out) return;
  int r = i / out, c = i % out;
  float s = 0.f;
  for (int k = 0; k < in; ++k) s += X[r * in + k] * W[k * out + c];
  Y[i] = s;
}

__global__ void bnrelu_batch_kernel(float* __restrict__ X, const float* __restrict__ g,
                                    const float* __restrict__ bb, int rows, int nch) {
  int c = blockIdx.x * blockDim.x + threadIdx.x;
  if (c >= nch) return;
  float s = 0.f, q = 0.f;
  for (int r = 0; r < rows; ++r) { float x = X[r * nch + c]; s += x; q += x * x; }
  float mean = s / rows;
  float var = q / rows - mean * mean;
  float sc = g[c] * rsqrtf(var + EPSV);
  float sh = bb[c] - mean * sc;
  for (int r = 0; r < rows; ++r) X[r * nch + c] = fmaxf(X[r * nch + c] * sc + sh, 0.f);
}

__global__ void fc3_kernel(const float* __restrict__ X, const float* __restrict__ W,
                           const float* __restrict__ bias, float* __restrict__ out) {
  int i = threadIdx.x;
  if (i >= 144) return;
  int r = i / 9, c = i % 9;
  float s = bias[c];
  for (int k = 0; k < 256; ++k) s += X[r * 256 + k] * W[k * 9 + c];
  out[i] = s;
}

// ---------------------------------------------------------------- launcher
extern "C" void kernel_launch(void* const* d_in, const int* in_sizes, int n_in,
                              void* d_out, int out_size, void* d_ws, size_t ws_size,
                              hipStream_t stream) {
  (void)in_sizes; (void)n_in; (void)out_size; (void)ws_size;
  const float* pc  = (const float*)d_in[0];
  const float* W1  = (const float*)d_in[1];
  const float* g1  = (const float*)d_in[2];
  const float* b1  = (const float*)d_in[3];
  const float* W2  = (const float*)d_in[4];
  const float* g2  = (const float*)d_in[5];
  const float* b2  = (const float*)d_in[6];
  const float* W3  = (const float*)d_in[7];
  const float* g3  = (const float*)d_in[8];
  const float* b3  = (const float*)d_in[9];
  const float* Wf1 = (const float*)d_in[10];
  const float* gf1 = (const float*)d_in[11];
  const float* bf1 = (const float*)d_in[12];
  const float* Wf2 = (const float*)d_in[13];
  const float* gf2 = (const float*)d_in[14];
  const float* bf2 = (const float*)d_in[15];
  const float* Wf3 = (const float*)d_in[16];
  const float* bf3 = (const float*)d_in[17];
  float* out = (float*)d_out;

  char* p = (char*)d_ws;
  int*   idx   = (int*)p;    p += (size_t)655360 * 4;
  float* sq    = (float*)p;  p += (size_t)32768 * 4;
  float* bn1   = (float*)p;  p += (size_t)4 * 64 * 4;
  float* bn2   = (float*)p;  p += (size_t)4 * 128 * 4;
  float* bn3   = (float*)p;  p += (size_t)4 * 1024 * 4;
  float* h2max = (float*)p;  p += (size_t)32768 * 128 * 4;
  float* h3max = (float*)p;  p += (size_t)16 * 1024 * 4;
  float* hA    = (float*)p;  p += (size_t)16 * 512 * 4;
  float* hB    = (float*)p;  p += (size_t)16 * 256 * 4;

  zero_kernel<<<1, 128, 0, stream>>>(bn1, 128);
  zero_kernel<<<1, 256, 0, stream>>>(bn2, 256);
  zero_kernel<<<8, 256, 0, stream>>>(bn3, 2048);
  zero_kernel<<<64, 256, 0, stream>>>(h3max, 16384);

  sq_kernel<<<128, 256, 0, stream>>>(pc, sq);
  knn_kernel<<<32768, 256, 0, stream>>>(pc, sq, idx);

  bn1_stats_kernel<<<2560, 256, 0, stream>>>(pc, idx, W1, bn1);
  bn_finalize_kernel<<<1, 64, 0, stream>>>(bn1, g1, b1, 64, 1.0f / 655360.0f);

  conv2_kernel<<<8192, 256, 0, stream>>>(pc, idx, W1, bn1, W2, bn2, h2max, 0);
  bn_finalize_kernel<<<1, 128, 0, stream>>>(bn2, g2, b2, 128, 1.0f / 655360.0f);
  conv2_kernel<<<8192, 256, 0, stream>>>(pc, idx, W1, bn1, W2, bn2, h2max, 1);

  conv3_kernel<<<dim3(512, 8), 256, 0, stream>>>(h2max, W3, bn3, h3max, 0);
  bn_finalize_kernel<<<4, 256, 0, stream>>>(bn3, g3, b3, 1024, 1.0f / 32768.0f);
  conv3_kernel<<<dim3(512, 8), 256, 0, stream>>>(h2max, W3, bn3, h3max, 1);

  fc_kernel<<<32, 256, 0, stream>>>(h3max, Wf1, hA, 16, 1024, 512);
  bnrelu_batch_kernel<<<2, 256, 0, stream>>>(hA, gf1, bf1, 16, 512);
  fc_kernel<<<16, 256, 0, stream>>>(hA, Wf2, hB, 16, 512, 256);
  bnrelu_batch_kernel<<<1, 256, 0, stream>>>(hB, gf2, bf2, 16, 256);
  fc3_kernel<<<1, 160, 0, stream>>>(hB, Wf3, bf3, out);
}
